// gLinear_11252814315777
// MI455X (gfx1250) — matrix operand order repaired and checked by reading, not compiled, not run on hardware
//
#include <hip/hip_runtime.h>
#include <hip/hip_bf16.h>

typedef __attribute__((ext_vector_type(16))) __bf16 v16bf;
typedef __attribute__((ext_vector_type(8)))  float  v8f;

#define B_ 128
#define V_ 1024
#define I_ 512
#define O_ 512
#define G_ 16

#define KC 32     // K-chunk staged in LDS per iteration (one bf16 WMMA K-step)
#define OT 128    // O columns per workgroup
#define SBF 40    // padded LDS row stride in bf16 elems (80 B: 16B-aligned rows, conflict-free)

// ---- packed f32x2 -> bf16x2 RNE; hi-f32 reconstructed from the packed bits ----
__device__ __forceinline__ unsigned pack2_bf16_rne(float a, float b) {
    __hip_bfloat162 h = __float22bfloat162_rn(make_float2(a, b));
    unsigned u;
    __builtin_memcpy(&u, &h, 4);
    return u;
}

// Split float4 into packed bf16 hi (RNE) and bf16 lo (residual, RNE).
__device__ __forceinline__ void split4(const float4 f, uint2& hi, uint2& lo) {
    hi.x = pack2_bf16_rne(f.x, f.y);
    hi.y = pack2_bf16_rne(f.z, f.w);
    const float hx = __uint_as_float(hi.x << 16);
    const float hy = __uint_as_float(hi.x & 0xFFFF0000u);
    const float hz = __uint_as_float(hi.y << 16);
    const float hw = __uint_as_float(hi.y & 0xFFFF0000u);
    lo.x = pack2_bf16_rne(f.x - hx, f.y - hy);
    lo.y = pack2_bf16_rne(f.z - hz, f.w - hw);
}

union Frag { uint4 q[2]; v16bf v; };

// ---- prep: split w [G,O,I] f32 into bf16 hi/lo planes in workspace (runs per launch) ----
__global__ __launch_bounds__(256) void split_w_kernel(
    const float* __restrict__ w,
    unsigned short* __restrict__ wsH,
    unsigned short* __restrict__ wsL)
{
    const size_t i4 = ((size_t)blockIdx.x * 256 + threadIdx.x) * 4;
    const float4 f = *(const float4*)(w + i4);
    uint2 h, l;
    split4(f, h, l);
    *(uint2*)(wsH + i4) = h;
    *(uint2*)(wsL + i4) = l;
}

template <bool PRESPLIT>
__global__ __launch_bounds__(256, 1) void glinear_wmma_bf16x3(
    const float*          __restrict__ x,      // [B, V, I]
    const int*            __restrict__ gidx,   // [V]
    const float*          __restrict__ w,      // [G, O, I] (used when !PRESPLIT)
    const unsigned short* __restrict__ wsH,    // [G, O, I] bf16 hi (PRESPLIT)
    const unsigned short* __restrict__ wsL,    // [G, O, I] bf16 lo (PRESPLIT)
    const float*          __restrict__ bias,   // [G, O]
    float*                __restrict__ out)    // [B, V, O]
{
    __shared__ __align__(16) unsigned short AsH[B_ * SBF]; // x chunk hi: 128 (b) x 32 (k)
    __shared__ __align__(16) unsigned short AsL[B_ * SBF];
    __shared__ __align__(16) unsigned short BsH[OT * SBF]; // w chunk hi: 128 (o) x 32 (k)
    __shared__ __align__(16) unsigned short BsL[OT * SBF];

    const int oy = blockIdx.x;          // O block: [oy*128, oy*128+128)
    const int v  = blockIdx.y;
    const int g  = gidx[v];

    const int tid  = threadIdx.x;
    const int wave = tid >> 5;          // 0..7 -> B-tile index
    const int lane = tid & 31;
    const int n    = lane & 15;         // M row (A) / N column (B,C,D) within tile
    const int hi16 = lane >> 4;
    const int kbA  = hi16 << 3;         // A frag: elems [kbA..+7] and [kbA+16..+23]
    const int kbB  = hi16 << 3;         // B frag: elems [kbB..+7] and [kbB+16..+23]

    // loader geometry: slot is constant per thread; rows advance by 32 per step
    const int lrow0 = tid >> 3;         // 0..31
    const int slot  = tid & 7;          // x: float4 slot 0..7 | w: H(0..3)/L(4..7) uint4 slot
    const int wsel  = slot >> 2;        // 0 -> hi plane, 1 -> lo plane
    const int wslot = slot & 3;         // uint4 slot within 32-bf16 row chunk

    const float* __restrict__ xcol = x + (size_t)v * I_;
    const size_t wbase = ((size_t)g * O_ + (size_t)oy * OT) * I_;
    const float* __restrict__ wrow = w + wbase;
    const unsigned short* __restrict__ wprow = (wsel ? wsL : wsH) + wbase;

    v8f acc[8];
    const v8f vzero = {0.f, 0.f, 0.f, 0.f, 0.f, 0.f, 0.f, 0.f};
#pragma unroll
    for (int t = 0; t < 8; ++t) acc[t] = vzero;

    // ---- software pipeline: preload chunk 0 into registers ----
    float4 rx[4];
    float4 rwf[4];   // !PRESPLIT path
    uint4  rwq[4];   // PRESPLIT path
#pragma unroll
    for (int i = 0; i < 4; ++i) {
        const int row = lrow0 + i * 32;
        rx[i] = *(const float4*)(xcol + (size_t)row * (V_ * I_) + slot * 4);
        if constexpr (PRESPLIT)
            rwq[i] = *(const uint4*)(wprow + (size_t)row * I_ + wslot * 8);
        else
            rwf[i] = *(const float4*)(wrow + (size_t)row * I_ + slot * 4);
    }

#pragma unroll 1
    for (int k0 = 0; k0 < I_; k0 += KC) {
        __syncthreads();   // previous chunk's LDS reads complete
#pragma unroll
        for (int i = 0; i < 4; ++i) {
            const int row = lrow0 + i * 32;
            uint2 h, l;
            split4(rx[i], h, l);
            *(uint2*)&AsH[row * SBF + slot * 4] = h;
            *(uint2*)&AsL[row * SBF + slot * 4] = l;
            if constexpr (PRESPLIT) {
                unsigned short* dst = wsel ? BsL : BsH;
                *(uint4*)&dst[row * SBF + wslot * 8] = rwq[i];
            } else {
                split4(rwf[i], h, l);
                *(uint2*)&BsH[row * SBF + slot * 4] = h;
                *(uint2*)&BsL[row * SBF + slot * 4] = l;
            }
        }
        __syncthreads();   // chunk visible to all waves

        // issue next chunk's global loads now; they land while WMMAs run
        const int kn = k0 + KC;
        if (kn < I_) {
#pragma unroll
            for (int i = 0; i < 4; ++i) {
                const int row = lrow0 + i * 32;
                rx[i] = *(const float4*)(xcol + (size_t)row * (V_ * I_) + kn + slot * 4);
                if constexpr (PRESPLIT)
                    rwq[i] = *(const uint4*)(wprow + (size_t)row * I_ + kn + wslot * 8);
                else
                    rwf[i] = *(const float4*)(wrow + (size_t)row * I_ + kn + slot * 4);
                __builtin_prefetch(xcol + (size_t)row * (V_ * I_) + kn + KC + slot * 4, 0, 0);
            }
        }

        // ---- compute: one bf16 K=32 step, 8 O-tiles x 3 split-products ----
        const int ar = (wave * 16 + n) * SBF;
        Frag aH, aL;
        aH.q[0] = *(const uint4*)&AsH[ar + kbA];
        aH.q[1] = *(const uint4*)&AsH[ar + kbA + 16];
        aL.q[0] = *(const uint4*)&AsL[ar + kbA];
        aL.q[1] = *(const uint4*)&AsL[ar + kbA + 16];
#pragma unroll
        for (int t = 0; t < 8; ++t) {
            const int br = (t * 16 + n) * SBF;
            Frag bH, bL;
            bH.q[0] = *(const uint4*)&BsH[br + kbB];
            bH.q[1] = *(const uint4*)&BsH[br + kbB + 16];
            bL.q[0] = *(const uint4*)&BsL[br + kbB];
            bL.q[1] = *(const uint4*)&BsL[br + kbB + 16];
            acc[t] = __builtin_amdgcn_wmma_f32_16x16x32_bf16(
                false, aH.v, false, bH.v, (short)0, acc[t], false, false);
            acc[t] = __builtin_amdgcn_wmma_f32_16x16x32_bf16(
                false, aL.v, false, bH.v, (short)0, acc[t], false, false);
            acc[t] = __builtin_amdgcn_wmma_f32_16x16x32_bf16(
                false, aH.v, false, bL.v, (short)0, acc[t], false, false);
        }
    }

    // ---- epilogue: bias add + store (C/D layout: VGPR r -> M = r or r+8) ----
    const float* __restrict__ bptr = bias + (size_t)g * O_;
    const int rowOff = hi16 << 3;   // 0 or 8
#pragma unroll
    for (int t = 0; t < 8; ++t) {
        const int o    = oy * OT + t * 16 + n;
        const float bv = bptr[o];
#pragma unroll
        for (int r = 0; r < 8; ++r) {
            const int b = wave * 16 + rowOff + r;
            out[((size_t)b * V_ + v) * O_ + o] = acc[t][r] + bv;
        }
    }
}

extern "C" void kernel_launch(void* const* d_in, const int* in_sizes, int n_in,
                              void* d_out, int out_size, void* d_ws, size_t ws_size,
                              hipStream_t stream) {
    const float* x    = (const float*)d_in[0];
    const int*   gidx = (const int*)  d_in[1];
    const float* w    = (const float*)d_in[2];
    const float* bias = (const float*)d_in[3];
    float*       out  = (float*)d_out;

    dim3 grid(O_ / OT, V_);   // oy fastest: same-v workgroups co-scheduled -> x hits L2

    const size_t wElems = (size_t)G_ * O_ * I_;          // 4.19M
    const size_t need   = wElems * 2 /*hi+lo*/ * sizeof(unsigned short); // ~16.8 MB

    if (ws_size >= need) {
        unsigned short* wsH = (unsigned short*)d_ws;
        unsigned short* wsL = wsH + wElems;
        split_w_kernel<<<(unsigned)(wElems / (4 * 256)), 256, 0, stream>>>(w, wsH, wsL);
        glinear_wmma_bf16x3<true><<<grid, 256, 0, stream>>>(
            x, gidx, w, wsH, wsL, bias, out);
    } else {
        glinear_wmma_bf16x3<false><<<grid, 256, 0, stream>>>(
            x, gidx, w, nullptr, nullptr, bias, out);
    }
}
